// ConvCRF_80092550135932
// MI455X (gfx1250) — matrix operand order repaired and checked by reading, not compiled, not run on hardware
//
#include <hip/hip_runtime.h>

// ---------------------------------------------------------------------------
// ConvCRF mean-field on MI455X (gfx1250).
// Message passing expressed as banded GEMMs on v_wmma_f32_16x16x32_bf16:
//   D[c, x(16)] += A[c, k(32)] * B[k, x]   per vertical tap offset dy (7x),
// A = 16-channel slab of a padded q row (bf16), B = per-tile banded weight
// matrix prebuilt in the exact per-lane WMMA B layout. Whole working set
// (~83 MB) is L2-resident (192 MB), so 5 iterations run out of L2.
// ---------------------------------------------------------------------------

typedef __attribute__((ext_vector_type(16))) __bf16 v16bf;
typedef __attribute__((ext_vector_type(8)))  __bf16 v8bf;
typedef __attribute__((ext_vector_type(8)))  float  v8f;

#define BB    2
#define CC    29
#define CP    32          // channel planes padded to 32 (planes 29..31 stay zero)
#define HH    256
#define WW    256
#define PADQ  3
#define QROWS 262         // 256 + 2*3
#define QCOLS 272         // >= 256+3+ (K up to 31 -> col 271), 16B aligned rows
#define NIT   5

static __device__ __forceinline__ unsigned short f2bf(float f) {
  union { float f; unsigned u; } v; v.f = f;
  unsigned u = v.u + 0x7fffu + ((v.u >> 16) & 1u);   // round-to-nearest-even
  return (unsigned short)(u >> 16);
}

// ---------------------------------------------------------------------------
// Zero the two padded q ping-pong buffers (halo + pad channel planes must be 0).
// ---------------------------------------------------------------------------
__global__ void crf_zero_kernel(uint4* __restrict__ p, long n16) {
  long i = (long)blockIdx.x * blockDim.x + threadIdx.x;
  if (i < n16) { uint4 z; z.x = 0u; z.y = 0u; z.z = 0u; z.w = 0u; p[i] = z; }
}

// ---------------------------------------------------------------------------
// q0 = softmax(-unary) over C=29, stored bf16 into padded planar q buffer.
// ---------------------------------------------------------------------------
__global__ void crf_initq_kernel(const float* __restrict__ unary,
                                 unsigned short* __restrict__ q) {
  int i = blockIdx.x * blockDim.x + threadIdx.x;     // B*H*W threads
  int x = i & 255, y = (i >> 8) & 255, b = i >> 16;
  const long hw = (long)HH * WW;
  const float* u = unary + (long)b * CC * hw + (long)y * WW + x;
  float l[CC];
  float m = -1e30f;
  #pragma unroll
  for (int c = 0; c < CC; ++c) { l[c] = -u[c * hw]; m = fmaxf(m, l[c]); }
  float s = 0.f;
  #pragma unroll
  for (int c = 0; c < CC; ++c) { l[c] = __expf(l[c] - m); s += l[c]; }
  float inv = 1.0f / s;
  const long plane = (long)QROWS * QCOLS;
  unsigned short* qp = q + (long)b * CP * plane + (long)(y + PADQ) * QCOLS + (x + PADQ);
  #pragma unroll
  for (int c = 0; c < CC; ++c) qp[c * plane] = f2bf(l[c] * inv);
}

// ---------------------------------------------------------------------------
// Build the banded weight tensor directly in per-lane WMMA B-operand layout:
//   bw[((b*7+dy)*256+y)*16+xt][lane][16 bf16], lane n -> pixel n&15,
//   K range n<16 ? 0..15 : 16..31 (ascending within the 32B).
//   B[k, pix] = (3 + 10*exp(-|dI|^2/200)) * sw[dy][k-pix], band 0<=k-pix<=6.
// One thread per lane record; each computes 16 bf16 slots and stores 32B.
// ---------------------------------------------------------------------------
__global__ void crf_weights_kernel(const float* __restrict__ img,
                                   unsigned short* __restrict__ bw) {
  int t = blockIdx.x * blockDim.x + threadIdx.x;     // B*7*256*16*32 threads
  int lane = t & 31;
  int xt   = (t >> 5) & 15;
  int y    = (t >> 9) & 255;
  int r    = t >> 17;
  int dy   = r % 7;
  int b    = r / 7;
  int p    = lane & 15;
  int half = lane >> 4;
  int x    = (xt << 4) | p;

  // separable 7-tap gaussian, sigma = 3
  float g[7]; float gs = 0.f;
  #pragma unroll
  for (int i = 0; i < 7; ++i) { float d = (float)(i - 3); g[i] = __expf(-d * d * (1.0f / 18.0f)); gs += g[i]; }
  float norm = 1.0f / (gs * gs);

  const long hw = (long)HH * WW;
  const float* ic = img + (long)b * 3 * hw + (long)y * WW + x;
  float c0 = ic[0], c1 = ic[hw], c2 = ic[2 * hw];

  unsigned short out[16];
  #pragma unroll
  for (int j = 0; j < 16; ++j) {
    int kk = (half << 4) | j;
    int dx = kk - p;
    float w = 0.f;
    if (dx >= 0 && dx <= 6) {
      int ny = y + dy - 3, nx = x + dx - 3;
      if (ny >= 0 && ny < HH && nx >= 0 && nx < WW) {
        const float* in_ = img + (long)b * 3 * hw + (long)ny * WW + nx;
        float d0 = in_[0] - c0, d1 = in_[hw] - c1, d2 = in_[2 * hw] - c2;
        float dsq = d0 * d0 + d1 * d1 + d2 * d2;
        float sw = g[dy] * g[dx] * norm;
        w = sw * (3.0f + 10.0f * __expf(-dsq * (1.0f / 200.0f)));
      }
    }
    out[j] = f2bf(w);
  }
  unsigned short* dst = bw + (long)t * 16;           // t is the lane-linear index
  ((uint4*)dst)[0] = *(const uint4*)&out[0];
  ((uint4*)dst)[1] = *(const uint4*)&out[8];
}

// ---------------------------------------------------------------------------
// One mean-field iteration. One wave per (b, y, 16-pixel x-tile):
//   14x v_wmma_f32_16x16x32_bf16 (7 dy x 2 channel tiles) -> pairwise,
//   then 29-way softmax via per-lane partials + shfl_xor(16).
// ---------------------------------------------------------------------------
__global__ void __launch_bounds__(256, 1)
crf_iter_kernel(const float* __restrict__ unary,
                const unsigned short* __restrict__ bw,
                const unsigned short* __restrict__ qsrc,
                unsigned short* __restrict__ qdst,
                float* __restrict__ outp,
                int final_iter) {
  int wave = (blockIdx.x << 3) | (threadIdx.x >> 5);
  int lane = threadIdx.x & 31;
  int xt = wave & 15;
  int y  = (wave >> 4) & 255;
  int b  = wave >> 12;
  int x0 = xt << 4;
  int p    = lane & 15;
  int half = lane >> 4;
  int m    = lane & 15;            // A-row (channel within tile) for this lane
  int klo  = half ? 16 : 0;        // A K-halves: contiguous k = 16*half + s (matches B)

  const long plane = (long)QROWS * QCOLS;
  const long hw = (long)HH * WW;
  const unsigned short* qb = qsrc + (long)b * CP * plane;

  v8f acc0 = {0.f, 0.f, 0.f, 0.f, 0.f, 0.f, 0.f, 0.f};
  v8f acc1 = {0.f, 0.f, 0.f, 0.f, 0.f, 0.f, 0.f, 0.f};

  #pragma unroll
  for (int dy = 0; dy < 7; ++dy) {
    int row = y + dy;              // padded-row index of image row (y+dy-3)
    const unsigned short* qr0 = qb + (long)m * plane + (long)row * QCOLS + x0;
    const unsigned short* qr1 = qb + (long)(16 + m) * plane + (long)row * QCOLS + x0;
    union { v16bf v; v8bf h[2]; } A0, A1;
    A0.h[0] = *(const v8bf*)(qr0 + klo);
    A0.h[1] = *(const v8bf*)(qr0 + klo + 8);
    A1.h[0] = *(const v8bf*)(qr1 + klo);
    A1.h[1] = *(const v8bf*)(qr1 + klo + 8);

    const unsigned short* bp =
        bw + ((((long)(b * 7 + dy) * 256 + y) * 16 + xt) * 32 + lane) * 16;
    v16bf Bv = *(const v16bf*)bp;
    if (dy < 6) __builtin_prefetch(bp + (1L << 21), 0, 1);  // next dy slab

    acc0 = __builtin_amdgcn_wmma_f32_16x16x32_bf16(false, A0.v, false, Bv,
                                                   (short)0, acc0, false, false);
    acc1 = __builtin_amdgcn_wmma_f32_16x16x32_bf16(false, A1.v, false, Bv,
                                                   (short)0, acc1, false, false);
  }

  // D layout: lane -> pixel p = lane&15; rows = half? 8..15 : 0..7.
  int c0 = half ? 8 : 0;           // channels for acc0 components
  int c1 = 16 + c0;                // channels for acc1 components
  const float* ub = unary + (long)b * CC * hw + (long)y * WW + (x0 + p);

  float l[16];
  #pragma unroll
  for (int j = 0; j < 8; ++j) {
    l[j] = acc0[j] - ub[(c0 + j) * hw];
    int cc = c1 + j;
    int cu = cc < CC ? cc : (CC - 1);                // clamp to keep load in-bounds
    float v = acc1[j] - ub[cu * hw];
    l[8 + j] = (cc < CC) ? v : -1e30f;               // mask padded channels 29..31
  }
  float mx = -1e30f;
  #pragma unroll
  for (int j = 0; j < 16; ++j) mx = fmaxf(mx, l[j]);
  mx = fmaxf(mx, __shfl_xor(mx, 16, 32));
  float s = 0.f;
  #pragma unroll
  for (int j = 0; j < 16; ++j) { l[j] = __expf(l[j] - mx); s += l[j]; }
  s += __shfl_xor(s, 16, 32);
  float inv = 1.0f / s;

  if (!final_iter) {
    unsigned short* qo = qdst + (long)b * CP * plane +
                         (long)(y + PADQ) * QCOLS + (x0 + p + PADQ);
    #pragma unroll
    for (int j = 0; j < 8; ++j) qo[(c0 + j) * plane] = f2bf(l[j] * inv);
    #pragma unroll
    for (int j = 0; j < 8; ++j) {
      int cc = c1 + j;
      if (cc < CC) qo[cc * plane] = f2bf(l[8 + j] * inv);
    }
  } else {
    float* op = outp + (long)b * CC * hw + (long)y * WW + (x0 + p);
    #pragma unroll
    for (int j = 0; j < 8; ++j) op[(c0 + j) * hw] = l[j] * inv;
    #pragma unroll
    for (int j = 0; j < 8; ++j) {
      int cc = c1 + j;
      if (cc < CC) op[cc * hw] = l[8 + j] * inv;
    }
  }
}

// ---------------------------------------------------------------------------
extern "C" void kernel_launch(void* const* d_in, const int* in_sizes, int n_in,
                              void* d_out, int out_size, void* d_ws, size_t ws_size,
                              hipStream_t stream) {
  (void)in_sizes; (void)n_in; (void)out_size; (void)ws_size;
  const float* unary = (const float*)d_in[0];
  const float* img   = (const float*)d_in[1];

  char* ws = (char*)d_ws;
  const long plane  = (long)QROWS * QCOLS;
  const long qbytes = (long)BB * CP * plane * 2;                 // ~9.1 MB each
  unsigned short* q0 = (unsigned short*)ws;
  unsigned short* q1 = (unsigned short*)(ws + qbytes);
  unsigned short* bw = (unsigned short*)(ws + 2 * qbytes);       // ~58.7 MB

  // 1) zero both q buffers (halo + padded channel planes must be zero)
  long n16 = (2 * qbytes) / 16;
  crf_zero_kernel<<<(int)((n16 + 255) / 256), 256, 0, stream>>>((uint4*)ws, n16);

  // 2) q0 = softmax(-unary)
  crf_initq_kernel<<<(BB * HH * WW) / 256, 256, 0, stream>>>(unary, q0);

  // 3) combined spatial+bilateral weights, prebuilt in WMMA B-operand layout
  crf_weights_kernel<<<(BB * 7 * 256 * 16 * 32) / 256, 256, 0, stream>>>(img, bw);

  // 4) five mean-field iterations (ping-pong), last writes f32 to d_out
  unsigned short* src = q0;
  unsigned short* dst = q1;
  for (int it = 0; it < NIT; ++it) {
    int fin = (it == NIT - 1);
    crf_iter_kernel<<<(BB * HH * 16) / 8, 256, 0, stream>>>(
        unary, bw, src, dst, (float*)d_out, fin);
    unsigned short* t = src; src = dst; dst = t;
  }
}
